// Det_AllHist_GLM_30571577213088
// MI455X (gfx1250) — compile-verified, matrix operand order checked by reading, not run on hardware
//
#include <hip/hip_runtime.h>
#include <hip/hip_bf16.h>

#define T_NOC 200
#define SUBC 20
#define E_NOC 2000
#define I_NOC 500
#define T_DATAC 10000
#define COS_NOC 17
#define PI_F 3.14159265358979f

// padded column count (SUB=20 -> 32 so two 16-wide WMMA N-tiles cover it)
#define NPAD 32
#define NCH_E 63   // ceil(2000/32)  (62 full + 1 tail)
#define NCH_I 16   // ceil(500/32)   (15 full + 1 tail)
#define NTILES_M (T_DATAC / 16)   // 625 exactly

typedef __attribute__((ext_vector_type(16))) __bf16 v16bf;
typedef __attribute__((ext_vector_type(8)))  float  v8f;

// ---- workspace layout (bytes) ----
#define OFF_SYNE 0u                                  // [10000][32] f32
#define OFF_SYNI (OFF_SYNE + T_DATAC*NPAD*4u)        // [10000][32] f32
#define OFF_FILT (OFF_SYNI + T_DATAC*NPAD*4u)        // [10000][32] f32 (includes Theta)
#define OFF_EK   (OFF_FILT + T_DATAC*NPAD*4u)        // [32][200] f32 (rows 20..31 zero)
#define OFF_IK   (OFF_EK + NPAD*T_NOC*4u)
#define OFF_SK   (OFF_IK + NPAD*T_NOC*4u)
#define OFF_HK   (OFF_SK + NPAD*T_NOC*4u)
#define OFF_BE   (OFF_HK + NPAD*T_NOC*4u)            // [63][2][32][16] bf16
#define OFF_BI   (OFF_BE + NCH_E*2u*32u*16u*2u)      // [16][2][32][16] bf16

__device__ __forceinline__ unsigned short f2bf_bits(float f) {
  union { float f; unsigned u; } x; x.f = f;
  unsigned u = x.u;
  unsigned r = (u + 0x7FFFu + ((u >> 16) & 1u)) >> 16;  // round-to-nearest-even
  return (unsigned short)r;
}

// ---------------------------------------------------------------------------
// Stage 1: build e/i synaptic kernels, spike kernel, history kernel.
// Writes padded [32][200] copies into ws and the 80x200 out_filters block.
// ---------------------------------------------------------------------------
__global__ void prep_filters(const float* __restrict__ Tau_syn,
                             const float* __restrict__ Delta_syn,
                             const float* __restrict__ W_syn,
                             const float* __restrict__ Tau_spk,
                             const float* __restrict__ W_spk,
                             const float* __restrict__ W_hist,
                             float* __restrict__ ek, float* __restrict__ ik,
                             float* __restrict__ sk, float* __restrict__ hk,
                             float* __restrict__ out_filters) {
  for (int i = threadIdx.x; i < NPAD * T_NOC; i += blockDim.x) {
    int s = i / T_NOC, j = i % T_NOC;
    float fe = 0.f, fi = 0.f, fs = 0.f, fh = 0.f;
    if (s < SUBC) {
      float tte = fmaxf((float)j - expf(Delta_syn[s*2+0]), 0.f) / expf(Tau_syn[s*2+0]);
      float tti = fmaxf((float)j - expf(Delta_syn[s*2+1]), 0.f) / expf(Tau_syn[s*2+1]);
      fe =  tte * expf(-tte) * expf(W_syn[s*2+0]);
      fi = -tti * expf(-tti) * expf(W_syn[s*2+1]);
      float tt = (float)j / expf(Tau_spk[s]);
      fs = tt * expf(-tt) * expf(W_spk[s]);
      float raw = 4.f * logf((float)j + 1.f);   // cos_scale*log(x+cos_shift)
      float acc = 0.f;
      #pragma unroll
      for (int c = 0; c < COS_NOC; ++c) {
        float phi = 0.5f * PI_F * (float)c;
        if (raw >= phi - PI_F && raw <= phi + PI_F)
          acc += W_hist[s*COS_NOC + c] * (0.5f * cosf(raw - phi) + 0.5f);
      }
      fh = acc;
      out_filters[(0*SUBC + s)*T_NOC + j] = fe;   // syn_filters: e rows 0..19
      out_filters[(1*SUBC + s)*T_NOC + j] = fi;   // i rows 20..39
      out_filters[(2*SUBC + s)*T_NOC + j] = fs;   // spk_kern rows 40..59
      out_filters[(3*SUBC + s)*T_NOC + j] = fh;   // hist_kern rows 60..79
    }
    ek[i] = fe; ik[i] = fi; sk[i] = fs; hk[i] = fh;
  }
}

// ---------------------------------------------------------------------------
// Stage 2: pack B = C_syn^T (K x N) into the bf16 WMMA B layout:
//   chunk c, n-tile nt, lane l, slot q :  n = nt*16 + (l&15),
//   k = c*32 + (l>=16 ? 16 : 0) + q.  Zero-padded beyond K / SUB.
// ---------------------------------------------------------------------------
__global__ void prep_B(const float* __restrict__ Ce, const float* __restrict__ Ci,
                       unsigned short* __restrict__ Be, unsigned short* __restrict__ Bi) {
  const int NE = NCH_E * 2 * 32 * 16;
  const int NI = NCH_I * 2 * 32 * 16;
  int idx = blockIdx.x * blockDim.x + threadIdx.x;
  if (idx < NE) {
    int q = idx & 15, l = (idx >> 4) & 31, nt = (idx >> 9) & 1, c = idx >> 10;
    int n = nt*16 + (l & 15);
    int k = c*32 + ((l >> 4) << 4) + q;
    float v = (n < SUBC && k < E_NOC) ? Ce[n*E_NOC + k] : 0.f;
    Be[idx] = f2bf_bits(v);
  } else if (idx < NE + NI) {
    int j = idx - NE;
    int q = j & 15, l = (j >> 4) & 31, nt = (j >> 9) & 1, c = j >> 10;
    int n = nt*16 + (l & 15);
    int k = c*32 + ((l >> 4) << 4) + q;
    float v = (n < SUBC && k < I_NOC) ? Ci[n*I_NOC + k] : 0.f;
    Bi[j] = f2bf_bits(v);
  }
}

// ---------------------------------------------------------------------------
// Stage 3: skinny spike GEMM with v_wmma_f32_16x16x32_bf16.
// One wave per 16-row time tile; blockIdx.y selects the e or i matrix.
// Spikes (0/1) and one-hot C_syn are exact in bf16; f32 accumulate is exact.
// Full K-chunks use unconditional float4 loads + hardware f32->bf16 cvt;
// the single ragged tail chunk is handled once outside the hot loop.
// ---------------------------------------------------------------------------
__global__ void spike_gemm(const float* __restrict__ Se, const float* __restrict__ Si,
                           const unsigned short* __restrict__ Be,
                           const unsigned short* __restrict__ Bi,
                           float* __restrict__ outE, float* __restrict__ outI) {
  int wid  = threadIdx.x >> 5;
  int lane = threadIdx.x & 31;
  int tile = blockIdx.x * 8 + wid;
  if (tile >= NTILES_M) return;                 // wave-uniform: EXEC stays all-1 for WMMA

  const float* S; const unsigned short* B; float* outp; int K, nch;
  if (blockIdx.y == 0) { S = Se; B = Be; outp = outE; K = E_NOC; nch = NCH_E; }
  else                 { S = Si; B = Bi; outp = outI; K = I_NOC; nch = NCH_I; }
  int nfull = nch - 1;                          // last chunk is ragged (K % 32 != 0)

  int row  = tile*16 + (lane & 15);             // A layout: lane&15 = M row
  int koff = (lane >> 4) << 4;                  // lanes 16..31 hold K=16..31
  const float* srow = S + (long)row * K;

  v8f acc0 = {0.f,0.f,0.f,0.f,0.f,0.f,0.f,0.f};
  v8f acc1 = acc0;

  for (int c = 0; c < nfull; ++c) {
    const float* sp = srow + c*32 + koff;
    float4 x0 = *(const float4*)(sp + 0);
    float4 x1 = *(const float4*)(sp + 4);
    float4 x2 = *(const float4*)(sp + 8);
    float4 x3 = *(const float4*)(sp + 12);
    if (c + 1 < nfull) __builtin_prefetch(sp + 32, 0, 0);   // next K chunk
    union { v16bf v; __bf16 b[16]; } a;
    a.b[0]  = (__bf16)x0.x; a.b[1]  = (__bf16)x0.y;
    a.b[2]  = (__bf16)x0.z; a.b[3]  = (__bf16)x0.w;
    a.b[4]  = (__bf16)x1.x; a.b[5]  = (__bf16)x1.y;
    a.b[6]  = (__bf16)x1.z; a.b[7]  = (__bf16)x1.w;
    a.b[8]  = (__bf16)x2.x; a.b[9]  = (__bf16)x2.y;
    a.b[10] = (__bf16)x2.z; a.b[11] = (__bf16)x2.w;
    a.b[12] = (__bf16)x3.x; a.b[13] = (__bf16)x3.y;
    a.b[14] = (__bf16)x3.z; a.b[15] = (__bf16)x3.w;
    v16bf b0 = *(const v16bf*)(B + ((c*2 + 0)*32 + lane)*16);
    v16bf b1 = *(const v16bf*)(B + ((c*2 + 1)*32 + lane)*16);
    acc0 = __builtin_amdgcn_wmma_f32_16x16x32_bf16(false, a.v, false, b0,
                                                   (short)0, acc0, false, false);
    acc1 = __builtin_amdgcn_wmma_f32_16x16x32_bf16(false, a.v, false, b1,
                                                   (short)0, acc1, false, false);
  }

  // ragged tail chunk (executed once): only slots q < rem (rem up to 20) are valid per lane
  {
    int c  = nfull;
    int k0 = c*32 + koff;
    int rem = K - k0;                       // lanes 16..31 may have rem <= 0
    union { v16bf v; __bf16 b[16]; } a;
    #pragma unroll
    for (int q = 0; q < 16; ++q)
      a.b[q] = (__bf16)((q < rem) ? srow[k0 + q] : 0.f);
    v16bf b0 = *(const v16bf*)(B + ((c*2 + 0)*32 + lane)*16);
    v16bf b1 = *(const v16bf*)(B + ((c*2 + 1)*32 + lane)*16);
    acc0 = __builtin_amdgcn_wmma_f32_16x16x32_bf16(false, a.v, false, b0,
                                                   (short)0, acc0, false, false);
    acc1 = __builtin_amdgcn_wmma_f32_16x16x32_bf16(false, a.v, false, b1,
                                                   (short)0, acc1, false, false);
  }

  // C/D layout: lanes 0-15 -> N=lane, M=j; lanes 16-31 -> N=lane-16, M=8+j
  int col   = lane & 15;
  int rbase = tile*16 + ((lane >> 4) ? 8 : 0);
  #pragma unroll
  for (int j = 0; j < 8; ++j) {
    outp[(rbase + j)*NPAD + col]      = acc0[j];
    outp[(rbase + j)*NPAD + 16 + col] = acc1[j];
  }
}

// ---------------------------------------------------------------------------
// Stage 4: strictly-causal depthwise FIR, filt[t][s] = Theta[s]
//   + sum_{d=1..200} ek[s][d-1]*synE[t-d][s] + ik[s][d-1]*synI[t-d][s]
// Kernels cached in LDS with pitch 201 (odd stride -> no bank conflicts).
// ---------------------------------------------------------------------------
__global__ void fir_filter(const float* __restrict__ synE, const float* __restrict__ synI,
                           const float* __restrict__ ekw, const float* __restrict__ ikw,
                           const float* __restrict__ Theta, float* __restrict__ filt) {
  __shared__ float ek[NPAD * 201];
  __shared__ float ik[NPAD * 201];
  for (int i = threadIdx.x; i < NPAD * T_NOC; i += blockDim.x) {
    int s = i / T_NOC, d = i % T_NOC;
    ek[s*201 + d] = ekw[i];
    ik[s*201 + d] = ikw[i];
  }
  __syncthreads();
  int s  = threadIdx.x & 31;
  int tg = threadIdx.x >> 5;
  int t0 = blockIdx.x * 64;
  float th = (s < SUBC) ? Theta[s] : 0.f;
  for (int tt = tg; tt < 64; tt += 8) {
    int t = t0 + tt;
    if (t >= T_DATAC) continue;
    float acc = th;
    int dmax = (t < T_NOC) ? t : T_NOC;
    for (int d = 1; d <= dmax; ++d) {
      int tp = t - d;
      acc += ek[s*201 + d - 1] * synE[tp*NPAD + s]
           + ik[s*201 + d - 1] * synI[tp*NPAD + s];
    }
    filt[t*NPAD + s] = acc;
  }
}

// ---------------------------------------------------------------------------
// Stage 5: the sequential GLM scan. One workgroup; spike history in a 256-deep
// LDS ring (ring[(t-d)&255] holds z[t-d]; untouched slots are the initial 0s).
// Thread (g,s): 25 lags (d = g*25+1 .. g*25+25) of subunit s, taps in VGPRs.
// ---------------------------------------------------------------------------
__global__ void scan_glm(const float* __restrict__ filt,
                         const float* __restrict__ khw,   // hist_kern [32][200]
                         const float* __restrict__ ksw,   // spk_kern  [32][200]
                         const float* __restrict__ Cden,
                         float* __restrict__ Z, float* __restrict__ P) {
  __shared__ float ring[256 * NPAD];
  __shared__ float partH[8 * NPAD];
  __shared__ float partV[8 * NPAD];
  __shared__ float hbuf[NPAD];
  __shared__ float vbuf[NPAD];
  __shared__ float cden[SUBC * SUBC];

  int tid = threadIdx.x;
  int s = tid & 31;
  int g = tid >> 5;
  for (int i = tid; i < 256 * NPAD; i += 256) ring[i] = 0.f;
  for (int i = tid; i < SUBC * SUBC; i += 256) cden[i] = Cden[i];

  float kh[25], ks[25];
  #pragma unroll
  for (int i = 0; i < 25; ++i) {
    kh[i] = khw[s*T_NOC + g*25 + i];   // tap for lag d = g*25+1+i (index d-1)
    ks[i] = ksw[s*T_NOC + g*25 + i];
  }
  __syncthreads();

  for (int t = 0; t < T_DATAC; ++t) {
    float h = 0.f, v = 0.f;
    #pragma unroll
    for (int i = 0; i < 25; ++i) {
      int d = g*25 + 1 + i;
      int pos = (t - d) & 255;
      float z = ring[pos*NPAD + s];
      h += z * kh[i];
      v += z * ks[i];
    }
    partH[g*NPAD + s] = h;
    partV[g*NPAD + s] = v;
    __syncthreads();
    if (tid < 32) {
      float hh = 0.f, vv = 0.f;
      #pragma unroll
      for (int gg = 0; gg < 8; ++gg) { hh += partH[gg*NPAD + s]; vv += partV[gg*NPAD + s]; }
      hbuf[s] = hh; vbuf[s] = vv;
    }
    __syncthreads();
    if (tid < 32) {
      if (s < SUBC) {
        float sv = hbuf[s] + filt[t*NPAD + s];
        float spk = 0.f;
        #pragma unroll
        for (int s2 = 0; s2 < SUBC; ++s2) spk += cden[s*SUBC + s2] * vbuf[s2];
        sv += spk;
        float z = (sv > 0.f) ? 1.f : 0.f;          // heaviside(x, 0)
        float p = 1.f / (1.f + expf(-sv));
        ring[(t & 255)*NPAD + s] = z;
        Z[t*SUBC + s] = z;
        P[t*SUBC + s] = p;
      } else {
        ring[(t & 255)*NPAD + s] = 0.f;
      }
    }
    __syncthreads();
  }
}

// ---------------------------------------------------------------------------
extern "C" void kernel_launch(void* const* d_in, const int* in_sizes, int n_in,
                              void* d_out, int out_size, void* d_ws, size_t ws_size,
                              hipStream_t stream) {
  const float* S_e       = (const float*)d_in[0];
  const float* S_i       = (const float*)d_in[1];
  const float* C_den     = (const float*)d_in[2];
  const float* C_syn_e   = (const float*)d_in[3];
  const float* C_syn_i   = (const float*)d_in[4];
  const float* Tau_syn   = (const float*)d_in[5];
  const float* Delta_syn = (const float*)d_in[6];
  const float* W_syn     = (const float*)d_in[7];
  const float* Tau_spk   = (const float*)d_in[8];
  const float* W_spk     = (const float*)d_in[9];
  const float* W_hist    = (const float*)d_in[10];
  const float* Theta     = (const float*)d_in[11];

  float* out     = (float*)d_out;
  float* Z       = out;                        // [10000][20]
  float* P       = out + T_DATAC * SUBC;       // [10000][20]
  float* filters = out + 2 * T_DATAC * SUBC;   // [80][200]

  char* ws = (char*)d_ws;
  float* synE = (float*)(ws + OFF_SYNE);
  float* synI = (float*)(ws + OFF_SYNI);
  float* filt = (float*)(ws + OFF_FILT);
  float* ek   = (float*)(ws + OFF_EK);
  float* ik   = (float*)(ws + OFF_IK);
  float* sk   = (float*)(ws + OFF_SK);
  float* hk   = (float*)(ws + OFF_HK);
  unsigned short* Be = (unsigned short*)(ws + OFF_BE);
  unsigned short* Bi = (unsigned short*)(ws + OFF_BI);

  prep_filters<<<1, 256, 0, stream>>>(Tau_syn, Delta_syn, W_syn, Tau_spk, W_spk,
                                      W_hist, ek, ik, sk, hk, filters);

  int nB = (NCH_E + NCH_I) * 2 * 32 * 16;
  prep_B<<<(nB + 255) / 256, 256, 0, stream>>>(C_syn_e, C_syn_i, Be, Bi);

  spike_gemm<<<dim3((NTILES_M + 7) / 8, 2), 256, 0, stream>>>(S_e, S_i, Be, Bi,
                                                              synE, synI);

  fir_filter<<<(T_DATAC + 63) / 64, 256, 0, stream>>>(synE, synI, ek, ik, Theta, filt);

  scan_glm<<<1, 256, 0, stream>>>(filt, hk, sk, C_den, Z, P);
}
